// Model_19181323944417
// MI455X (gfx1250) — matrix operand order repaired and checked by reading, not compiled, not run on hardware
//
#include <hip/hip_runtime.h>
#include <hip/hip_bf16.h>

#define NU   100000
#define NI   50000
#define NT   150000
#define DIM  64
#define EE   2000000LL
#define CLEPS 0.2f
#define TAU   0.5f

typedef __attribute__((ext_vector_type(16))) _Float16 v16h;
typedef __attribute__((ext_vector_type(8)))  float    v8f;

// ---------- helpers ----------
__device__ __forceinline__ float sgnf(float x){ return (x > 0.f) ? 1.f : ((x < 0.f) ? -1.f : 0.f); }

// order-preserving encode of f32 into u32 (for atomicMax over floats of any sign)
__device__ __forceinline__ unsigned enc_f32(float f){
  unsigned x = __float_as_uint(f);
  return (x >> 31) ? ~x : (x | 0x80000000u);
}
__device__ __forceinline__ float dec_f32(unsigned e){
  unsigned x = (e >> 31) ? (e & 0x7FFFFFFFu) : ~e;
  return __uint_as_float(x);
}

__device__ __forceinline__ float4 ld4(const float* p){ return *(const float4*)p; }

__device__ __forceinline__ void pack8(v16h& v, int off, float4 a, float4 b){
  v[off+0]=(_Float16)a.x; v[off+1]=(_Float16)a.y; v[off+2]=(_Float16)a.z; v[off+3]=(_Float16)a.w;
  v[off+4]=(_Float16)b.x; v[off+5]=(_Float16)b.y; v[off+6]=(_Float16)b.z; v[off+7]=(_Float16)b.w;
}

// ---------- generic fills ----------
__global__ void __launch_bounds__(256) k_zero(float* p, long long n){
  long long i = (long long)blockIdx.x * blockDim.x + threadIdx.x;
  if (i < n) p[i] = 0.f;
}
__global__ void __launch_bounds__(256) k_fill_u32(unsigned* p, long long n, unsigned v){
  long long i = (long long)blockIdx.x * blockDim.x + threadIdx.x;
  if (i < n) p[i] = v;
}

// ---------- degrees & normalization coefficients ----------
__global__ void __launch_bounds__(256) k_degree(const int* __restrict__ eu, const int* __restrict__ ei,
                                                float* du, float* di){
  long long e = (long long)blockIdx.x * blockDim.x + threadIdx.x;
  if (e >= EE) return;
  atomicAdd(&du[eu[e]], 1.f);
  atomicAdd(&di[ei[e]], 1.f);
}
__global__ void __launch_bounds__(256) k_invsqrt(float* c, long long n){
  long long i = (long long)blockIdx.x * blockDim.x + threadIdx.x;
  if (i < n) c[i] = rsqrtf(fmaxf(c[i], 1.f));
}

// ---------- init: concat embeddings into A and layer-sum S ----------
__global__ void __launch_bounds__(256) k_init(const float* __restrict__ ue, const float* __restrict__ ie,
                                              float* A, float* S){
  long long i = (long long)blockIdx.x * blockDim.x + threadIdx.x;
  if (i >= (long long)NT * DIM) return;
  float v = (i < (long long)NU * DIM) ? ue[i] : ie[i - (long long)NU * DIM];
  A[i] = v; S[i] = v;
}

// ---------- GCN propagation: edge x dim parallel, L2-resident atomics ----------
__global__ void __launch_bounds__(256) k_gcn(const float* __restrict__ cur, float* nxt,
                                             const int* __restrict__ eu, const int* __restrict__ ei,
                                             const float* __restrict__ cu, const float* __restrict__ ci){
  long long idx = (long long)blockIdx.x * blockDim.x + threadIdx.x;
  if (idx >= EE * DIM) return;
  int d = (int)(idx & (DIM - 1));
  long long e = idx >> 6;
  int u = eu[e], it = ei[e];
  float w = cu[u] * ci[it];
  float iv = cur[((size_t)(NU + it)) * DIM + d];
  float uv = cur[(size_t)u * DIM + d];
  atomicAdd(&nxt[(size_t)u * DIM + d],         w * iv);
  atomicAdd(&nxt[((size_t)(NU + it)) * DIM + d], w * uv);
}

// ---------- contrastive noise: one wave per row (wave32, 2 dims/lane) ----------
__global__ void __launch_bounds__(256) k_cl(float* emb, const float* __restrict__ nz, float* S){
  long long wid = ((long long)blockIdx.x * blockDim.x + threadIdx.x) >> 5;
  int lane = threadIdx.x & 31;
  if (wid >= NT) return;
  const float* nrow = nz + (size_t)wid * DIM;
  float n0 = nrow[lane], n1 = nrow[lane + 32];
  float ss = n0 * n0 + n1 * n1;
  #pragma unroll
  for (int m = 16; m >= 1; m >>= 1) ss += __shfl_xor(ss, m, 32);
  float inv = 1.f / fmaxf(sqrtf(ss), 1e-12f);
  size_t off = (size_t)wid * DIM;
  float e0 = emb[off + lane], e1 = emb[off + lane + 32];
  e0 += sgnf(e0) * n0 * inv * CLEPS;
  e1 += sgnf(e1) * n1 * inv * CLEPS;
  emb[off + lane] = e0;       emb[off + lane + 32] = e1;
  S[off + lane]  += e0;       S[off + lane + 32]  += e1;
}

__global__ void __launch_bounds__(256) k_scale(const float* __restrict__ S, float* A, long long n, float f){
  long long i = (long long)blockIdx.x * blockDim.x + threadIdx.x;
  if (i < n) A[i] = S[i] * f;
}

// ---------- rank-former scores via WMMA: 16 edge dot-products per wave ----------
// A (16x32 f16): lane m (h=0) holds row m dims {0..7,16..23}; lane 16+m (h=1) holds {8..15,24..31}.
// B (32x16 f16): lane n (h=0) holds col n dims {0..7,16..23}; lane 16+n holds {8..15,24..31}.
// Two WMMAs cover D=64; C diagonal (M=N) lives in lanes 0..7 (c[lane]) and 24..31 (c[lane-24]).
__global__ void __launch_bounds__(256) k_score(const float* __restrict__ emb,
                                               const int* __restrict__ eu, const int* __restrict__ ei,
                                               float* __restrict__ s){
  long long grp  = ((long long)blockIdx.x * blockDim.x + threadIdx.x) >> 5;
  int lane = threadIdx.x & 31;
  long long base = grp * 16;
  if (base >= EE) return;
  int m = lane & 15, h = lane >> 4;
  const float* ur = emb + (size_t)eu[base + m] * DIM;
  const float* ir = emb + (size_t)(NU + ei[base + m]) * DIM;

  v16h a_lo, a_hi, b_lo, b_hi;
  // A operand (user rows)
  pack8(a_lo, 0, ld4(ur + 8*h),          ld4(ur + 8*h + 4));
  pack8(a_lo, 8, ld4(ur + 16 + 8*h),     ld4(ur + 16 + 8*h + 4));
  pack8(a_hi, 0, ld4(ur + 32 + 8*h),     ld4(ur + 32 + 8*h + 4));
  pack8(a_hi, 8, ld4(ur + 48 + 8*h),     ld4(ur + 48 + 8*h + 4));
  // B operand (item rows as columns)
  pack8(b_lo, 0, ld4(ir + 8*h),          ld4(ir + 8*h + 4));
  pack8(b_lo, 8, ld4(ir + 16 + 8*h),     ld4(ir + 16 + 8*h + 4));
  pack8(b_hi, 0, ld4(ir + 32 + 8*h),     ld4(ir + 32 + 8*h + 4));
  pack8(b_hi, 8, ld4(ir + 48 + 8*h),     ld4(ir + 48 + 8*h + 4));

  v8f c = {};
  c = __builtin_amdgcn_wmma_f32_16x16x32_f16(false, a_lo, false, b_lo, (short)0, c, false, false);
  c = __builtin_amdgcn_wmma_f32_16x16x32_f16(false, a_hi, false, b_hi, (short)0, c, false, false);

  int sel = (lane < 8) ? lane : (lane - 24);
  float v = c[0];
  #pragma unroll
  for (int k = 1; k < 8; ++k) v = (sel == k) ? (float)c[k] : v;
  if (lane < 8)        s[base + lane]        = v;   // RF_ALPHA = 1.0
  else if (lane >= 24) s[base + (lane - 16)] = v;
}

// ---------- segment max via order-encoded u32 atomicMax ----------
__global__ void __launch_bounds__(256) k_segmax(const float* __restrict__ s,
                                                const int* __restrict__ eu, const int* __restrict__ ei,
                                                unsigned* mu, unsigned* mi){
  long long e = (long long)blockIdx.x * blockDim.x + threadIdx.x;
  if (e >= EE) return;
  unsigned en = enc_f32(s[e]);
  atomicMax(&mu[eu[e]], en);
  atomicMax(&mi[ei[e]], en);
}

// ---------- fused softmax numerator + z + rec accumulation ----------
__global__ void __launch_bounds__(256) k_attn(const float* __restrict__ emb, const float* __restrict__ s,
                                              const int* __restrict__ eu, const int* __restrict__ ei,
                                              const unsigned* __restrict__ mu, const unsigned* __restrict__ mi,
                                              float* zu, float* zi, float* rec){
  long long idx = (long long)blockIdx.x * blockDim.x + threadIdx.x;
  if (idx >= EE * DIM) return;
  int d = (int)(idx & (DIM - 1));
  long long e = idx >> 6;
  int u = eu[e], it = ei[e];
  float sv = s[e];
  float pu = __expf(sv - dec_f32(mu[u]));
  float pi = __expf(sv - dec_f32(mi[it]));
  if (d == 0){ atomicAdd(&zu[u], pu); atomicAdd(&zi[it], pi); }
  atomicAdd(&rec[(size_t)u * DIM + d],          pu * emb[(size_t)(NU + it) * DIM + d]);
  atomicAdd(&rec[(size_t)(NU + it) * DIM + d],  pi * emb[(size_t)u * DIM + d]);
}

__global__ void __launch_bounds__(256) k_blend(float* emb, const float* __restrict__ rec,
                                               const float* __restrict__ zu, const float* __restrict__ zi){
  long long i = (long long)blockIdx.x * blockDim.x + threadIdx.x;
  if (i >= (long long)NT * DIM) return;
  long long row = i >> 6;
  float z = (row < NU) ? zu[row] : zi[row - NU];
  float r = rec[i] / fmaxf(z, 1e-9f);
  emb[i] = (1.f - TAU) * emb[i] + TAU * r;
}

// ---------- host orchestration ----------
extern "C" void kernel_launch(void* const* d_in, const int* in_sizes, int n_in,
                              void* d_out, int out_size, void* d_ws, size_t ws_size,
                              hipStream_t stream){
  const float* user_emb = (const float*)d_in[0];
  const float* item_emb = (const float*)d_in[1];
  const float* noise    = (const float*)d_in[2];
  const int*   edge_u   = (const int*)d_in[3];
  const int*   edge_i   = (const int*)d_in[4];

  float* ws   = (float*)d_ws;
  float* cu   = ws;                       // NU
  float* ci   = cu + NU;                  // NI
  float* bufA = ci + NI;                  // NT*DIM
  float* bufB = bufA + (size_t)NT * DIM;  // NT*DIM
  float* sbuf = bufB + (size_t)NT * DIM;  // EE
  unsigned* mu_e = (unsigned*)(sbuf + EE);// NU
  unsigned* mi_e = mu_e + NU;             // NI
  float* zu   = (float*)(mi_e + NI);      // NU
  float* zi   = zu + NU;                  // NI
  float* S    = (float*)d_out;            // layer-sum, later rec accumulator

  const int B = 256;
  auto g = [](long long n){ return (unsigned)((n + 255) / 256); };
  const long long ND = (long long)NT * DIM;

  // degrees -> cu, ci
  k_zero   <<<g(NU + NI), B, 0, stream>>>(cu, NU + NI);
  k_degree <<<g(EE),      B, 0, stream>>>(edge_u, edge_i, cu, ci);
  k_invsqrt<<<g(NU + NI), B, 0, stream>>>(cu, NU + NI);

  // emb0 -> bufA, S
  k_init<<<g(ND), B, 0, stream>>>(user_emb, item_emb, bufA, S);

  float* curp = bufA; float* nxtp = bufB;
  for (int l = 0; l < 3; ++l){
    k_zero<<<g(ND),        B, 0, stream>>>(nxtp, ND);
    k_gcn <<<g(EE * DIM),  B, 0, stream>>>(curp, nxtp, edge_u, edge_i, cu, ci);
    k_cl  <<<g((long long)NT * 32), B, 0, stream>>>(nxtp, noise + (size_t)l * NT * DIM, S);
    float* t = curp; curp = nxtp; nxtp = t;
  }

  float* embM = nxtp;                     // free ping-pong buffer
  k_scale<<<g(ND), B, 0, stream>>>(S, embM, ND, 0.25f);

  for (int r = 0; r < 2; ++r){
    k_score   <<<(unsigned)(EE / 16 / 8), B, 0, stream>>>(embM, edge_u, edge_i, sbuf);
    k_fill_u32<<<g(NU + NI), B, 0, stream>>>(mu_e, NU + NI, 0x00800000u); // enc(-FLT_MAX)
    k_segmax  <<<g(EE),      B, 0, stream>>>(sbuf, edge_u, edge_i, mu_e, mi_e);
    k_zero    <<<g(NU + NI), B, 0, stream>>>(zu, NU + NI);
    k_zero    <<<g(ND),      B, 0, stream>>>(S, ND);                      // rec
    k_attn    <<<g(EE * DIM),B, 0, stream>>>(embM, sbuf, edge_u, edge_i, mu_e, mi_e, zu, zi, S);
    k_blend   <<<g(ND),      B, 0, stream>>>(embM, S, zu, zi);
  }

  hipMemcpyAsync(d_out, embM, (size_t)NT * DIM * sizeof(float),
                 hipMemcpyDeviceToDevice, stream);
}
